// GCN_LP_46600395161977
// MI455X (gfx1250) — matrix operand order repaired and checked by reading, not compiled, not run on hardware
//
#include <hip/hip_runtime.h>
#include <hip/hip_bf16.h>
#include <math.h>

typedef __attribute__((ext_vector_type(16))) _Float16 v16h;
typedef __attribute__((ext_vector_type(8)))  _Float16 h8;
typedef __attribute__((ext_vector_type(8)))  float    v8f;
typedef __attribute__((ext_vector_type(4)))  _Float16 h4;
typedef __attribute__((ext_vector_type(4)))  float    f4;

__device__ __forceinline__ v16h cat16(h8 lo, h8 hi) {
    return __builtin_shufflevector(lo, hi, 0, 1, 2, 3, 4, 5, 6, 7,
                                           8, 9, 10, 11, 12, 13, 14, 15);
}

// ---------------------------------------------------------------- utilities
__global__ void zero_f32_kernel(float* __restrict__ p, long long n) {
    long long i = (long long)blockIdx.x * blockDim.x + threadIdx.x;
    long long stride = (long long)gridDim.x * blockDim.x;
    for (; i < n; i += stride) p[i] = 0.0f;
}

__global__ void degree_kernel(const int* __restrict__ src, const int* __restrict__ dst,
                              float* __restrict__ deg_out, float* __restrict__ deg_in, int E) {
    int e = blockIdx.x * blockDim.x + threadIdx.x;
    if (e >= E) return;
    atomicAdd(&deg_out[src[e]], 1.0f);
    atomicAdd(&deg_in[dst[e]], 1.0f);
}

__global__ void rsqrt_kernel(float* __restrict__ deg, int n) {
    int i = blockIdx.x * blockDim.x + threadIdx.x;
    if (i >= n) return;
    float v = deg[i];
    deg[i] = rsqrtf(fmaxf(v, 1.0f));
}

// ------------------------------------------------------- WMMA GEMM (GCN fc)
// Out[N,NOUT](f16) = A[N,128] @ W[128,NOUT](f32). 64-row macro tile per block.
// W staged transposed in LDS (K-contiguous per column) so B frags are 2x b128.
template <int NOUT, bool AHALF>
__global__ __launch_bounds__(256) void gcn_gemm_kernel(
    const void* __restrict__ Ain, const float* __restrict__ rowScale,
    const float* __restrict__ W, _Float16* __restrict__ Out, int Nrows) {
    constexpr int CW = NOUT / 16;   // column waves (8 or 4)
    constexpr int RG = 8 / CW;      // row groups (1 or 2)
    constexpr int MT = 4 / RG;      // 16-row tiles per wave

    __shared__ alignas(16) _Float16 As[64 * 136];
    __shared__ alignas(16) _Float16 Wt[NOUT * 136];

    const int tid = threadIdx.x;
    const int rb  = blockIdx.x * 64;

    // stage A (64x128 rows, f16, row-scaled) -- 8 halves per iteration
    for (int i = tid; i < 64 * 16; i += 256) {
        int r = i >> 4, c8 = (i & 15) * 8;
        int row = rb + r;
        h8 v = (h8)(_Float16)0.0f;
        if (row < Nrows) {
            if (AHALF) {
                v = *(const h8*)((const _Float16*)Ain + (size_t)row * 128 + c8);
            } else {
                const float* p = (const float*)Ain + (size_t)row * 128 + c8;
                f4 q0 = *(const f4*)p;
                f4 q1 = *(const f4*)(p + 4);
                float sc = rowScale[row];
#pragma unroll
                for (int j = 0; j < 4; ++j) {
                    v[j]     = (_Float16)(q0[j] * sc);
                    v[j + 4] = (_Float16)(q1[j] * sc);
                }
            }
        }
        *(h8*)(&As[r * 136 + c8]) = v;
    }
    // stage W transposed: Wt[n][k] (k contiguous)
    for (int i = tid; i < 128 * NOUT; i += 256) {
        int k = i / NOUT, n = i % NOUT;
        Wt[n * 136 + k] = (_Float16)W[i];
    }
    __syncthreads();

    const int wave = tid >> 5;
    const int lane = tid & 31;
    const int m = lane & 15, g = lane >> 4;
    const int cw = wave % CW, rg = wave / CW;
    const int n = m + cw * 16;

    // hoist B fragments for all 4 k-steps (K = ks*32 + 16g .. +15, contiguous)
    v16h bf[4];
#pragma unroll
    for (int ks = 0; ks < 4; ++ks) {
        const _Float16* bp = &Wt[n * 136 + ks * 32 + g * 16];
        bf[ks] = cat16(*(const h8*)bp, *(const h8*)(bp + 8));
    }

#pragma unroll
    for (int t = 0; t < MT; ++t) {
        const int mt = rg * MT + t;
        v8f acc = {};
#pragma unroll
        for (int ks = 0; ks < 4; ++ks) {
            const _Float16* ap = &As[(mt * 16 + m) * 136 + ks * 32 + g * 16];
            v16h a = cat16(*(const h8*)ap, *(const h8*)(ap + 8));
            acc = __builtin_amdgcn_wmma_f32_16x16x32_f16(false, a, false, bf[ks], (short)0,
                                                         acc, false, false);
        }
#pragma unroll
        for (int r = 0; r < 8; ++r) {
            int row = rb + mt * 16 + r + 8 * g;
            if (row < Nrows) Out[(size_t)row * NOUT + n] = (_Float16)acc[r];
        }
    }
}

// ------------------------------------------------ edge scatter-add (segment_sum)
template <int C>
__global__ void agg_kernel(const _Float16* __restrict__ t, const int* __restrict__ src,
                           const int* __restrict__ dst, float* __restrict__ agg, int E) {
    constexpr int TPE = C / 4;
    long long gid = (long long)blockIdx.x * blockDim.x + threadIdx.x;
    long long total = (long long)E * TPE;
    if (gid >= total) return;
    int e    = (int)(gid / TPE);
    int lane = (int)(gid % TPE);
    int s = src[e], d = dst[e];
    h4 v = *(const h4*)(t + (size_t)s * C + 4 * lane);
    float* p = agg + (size_t)d * C + 4 * lane;
    atomicAdd(p + 0, (float)v[0]);
    atomicAdd(p + 1, (float)v[1]);
    atomicAdd(p + 2, (float)v[2]);
    atomicAdd(p + 3, (float)v[3]);
}

__global__ void post1_kernel(const float* __restrict__ agg, const float* __restrict__ rs_out,
                             const float* __restrict__ rs_in, const float* __restrict__ b1,
                             _Float16* __restrict__ h1, int Nrows) {
    long long idx = (long long)blockIdx.x * blockDim.x + threadIdx.x;
    if (idx >= (long long)Nrows * 128) return;
    int i = (int)(idx >> 7), c = (int)(idx & 127);
    float v = agg[idx] * rs_in[i] + b1[c];
    v = fmaxf(v, 0.0f);
    h1[idx] = (_Float16)(v * rs_out[i]);
}

__global__ void post2_kernel(float* __restrict__ agg, const float* __restrict__ rs_in,
                             const float* __restrict__ b2, int Nrows) {
    long long idx = (long long)blockIdx.x * blockDim.x + threadIdx.x;
    if (idx >= (long long)Nrows * 64) return;
    int i = (int)(idx >> 6), c = (int)(idx & 63);
    agg[idx] = agg[idx] * rs_in[i] + b2[c];
}

// ---------------------------------------------------- edge MLP (WMMA, 16 edges/wave)
__global__ __launch_bounds__(256) void edge_mlp_kernel(
    const float* __restrict__ h2, const int* __restrict__ src, const int* __restrict__ dst,
    const float* __restrict__ Wp1, const float* __restrict__ bp1,
    const float* __restrict__ Wp2, const float* __restrict__ bp2,
    const float* __restrict__ Wp3, const float* __restrict__ bp3,
    float* __restrict__ out, int E) {
    __shared__ alignas(16) _Float16 sW1t[64 * 136];   // [n][k], k<128
    __shared__ alignas(16) _Float16 sW2t[32 * 136];   // [n][k], k<64
    __shared__ alignas(16) _Float16 zs[8][16 * 136];  // per-wave z scratch
    __shared__ float sB1[64], sB2[32], sW3[32];
    __shared__ float sBp3;

    const int tid = threadIdx.x;
    for (int i = tid; i < 128 * 64; i += 256) {       // Wp1 [128][64] -> [n][k]
        int k = i >> 6, n = i & 63;
        sW1t[n * 136 + k] = (_Float16)Wp1[i];
    }
    for (int i = tid; i < 64 * 32; i += 256) {        // Wp2 [64][32] -> [n][k]
        int k = i >> 5, n = i & 31;
        sW2t[n * 136 + k] = (_Float16)Wp2[i];
    }
    if (tid < 64) sB1[tid] = bp1[tid];
    if (tid < 32) { sB2[tid] = bp2[tid]; sW3[tid] = Wp3[tid]; }
    if (tid == 0) sBp3 = bp3[0];
    __syncthreads();

    const int wave = tid >> 5;
    const int lane = tid & 31;
    const int m = lane & 15, g = lane >> 4;
    const int eb = blockIdx.x * 128 + wave * 16;
    _Float16* zb = &zs[wave][0];

    int e = eb + m; if (e >= E) e = E - 1;            // clamp: keep EXEC all-ones
    const int s = src[e];
    const int d = dst[e];

    // ---- layer 1: z1[16,64] = relu(concat(h2[s],h2[d]) @ Wp1 + bp1)
    v8f acc1[4] = {};
#pragma unroll
    for (int ks = 0; ks < 4; ++ks) {
        // k-step lies entirely in src half (ks<2) or dst half (ks>=2)
        const float* rowp = (ks < 2) ? (h2 + (size_t)s * 64) : (h2 + (size_t)d * 64 - 64);
        const float* p0 = rowp + ks * 32 + 16 * g;
        const float* p1 = p0 + 8;
        f4 q0 = *(const f4*)p0;
        f4 q1 = *(const f4*)(p0 + 4);
        f4 q2 = *(const f4*)p1;
        f4 q3 = *(const f4*)(p1 + 4);
        v16h a;
#pragma unroll
        for (int j = 0; j < 4; ++j) {
            a[j]      = (_Float16)q0[j];
            a[j + 4]  = (_Float16)q1[j];
            a[j + 8]  = (_Float16)q2[j];
            a[j + 12] = (_Float16)q3[j];
        }
#pragma unroll
        for (int cn = 0; cn < 4; ++cn) {
            const _Float16* bp = &sW1t[(m + 16 * cn) * 136 + ks * 32 + g * 16];
            v16h b = cat16(*(const h8*)bp, *(const h8*)(bp + 8));
            acc1[cn] = __builtin_amdgcn_wmma_f32_16x16x32_f16(false, a, false, b, (short)0,
                                                              acc1[cn], false, false);
        }
    }
#pragma unroll
    for (int cn = 0; cn < 4; ++cn)
#pragma unroll
        for (int r = 0; r < 8; ++r) {
            int n = m + 16 * cn, mm = r + 8 * g;
            float v = fmaxf(acc1[cn][r] + sB1[n], 0.0f);
            zb[mm * 136 + n] = (_Float16)v;
        }
    asm volatile("s_wait_dscnt 0" ::: "memory");      // wave-local LDS handoff

    // ---- layer 2: z2[16,32] = relu(z1 @ Wp2 + bp2)
    v8f acc2[2] = {};
#pragma unroll
    for (int ks = 0; ks < 2; ++ks) {
        const _Float16* ap = &zb[m * 136 + ks * 32 + g * 16];
        v16h a = cat16(*(const h8*)ap, *(const h8*)(ap + 8));
#pragma unroll
        for (int cn = 0; cn < 2; ++cn) {
            const _Float16* bp = &sW2t[(m + 16 * cn) * 136 + ks * 32 + g * 16];
            v16h b = cat16(*(const h8*)bp, *(const h8*)(bp + 8));
            acc2[cn] = __builtin_amdgcn_wmma_f32_16x16x32_f16(false, a, false, b, (short)0,
                                                              acc2[cn], false, false);
        }
    }
    asm volatile("s_wait_dscnt 0" ::: "memory");      // WAR: layer-2 reads done
#pragma unroll
    for (int cn = 0; cn < 2; ++cn)
#pragma unroll
        for (int r = 0; r < 8; ++r) {
            int n = m + 16 * cn, mm = r + 8 * g;
            float v = fmaxf(acc2[cn][r] + sB2[n], 0.0f);
            zb[mm * 136 + n] = (_Float16)v;
        }
    asm volatile("s_wait_dscnt 0" ::: "memory");

    // ---- layer 3 + sigmoid: one edge per lane (lanes 0..15)
    if (lane < 16 && (eb + lane) < E) {
        float sum = sBp3;
#pragma unroll
        for (int n = 0; n < 32; ++n) sum += (float)zb[lane * 136 + n] * sW3[n];
        out[eb + lane] = 1.0f / (1.0f + expf(-sum));
    }
}

// ------------------------------------------------------------------ launcher
extern "C" void kernel_launch(void* const* d_in, const int* in_sizes, int n_in,
                              void* d_out, int out_size, void* d_ws, size_t ws_size,
                              hipStream_t stream) {
    const float* x   = (const float*)d_in[0];
    const int*   src = (const int*)d_in[1];
    const int*   dst = (const int*)d_in[2];
    const float* W1  = (const float*)d_in[3];
    const float* b1  = (const float*)d_in[4];
    const float* W2  = (const float*)d_in[5];
    const float* b2  = (const float*)d_in[6];
    const float* Wp1 = (const float*)d_in[7];
    const float* bp1 = (const float*)d_in[8];
    const float* Wp2 = (const float*)d_in[9];
    const float* bp2 = (const float*)d_in[10];
    const float* Wp3 = (const float*)d_in[11];
    const float* bp3 = (const float*)d_in[12];
    float* out = (float*)d_out;

    const int N = in_sizes[0] / 128;   // nodes
    const int E = in_sizes[1];         // edges

    auto align256 = [](size_t v) { return (v + 255) & ~(size_t)255; };
    char* base = (char*)d_ws;
    size_t off = 0;
    float* deg = (float*)(base + off); off += align256((size_t)2 * N * 4);
    _Float16* t1 = (_Float16*)(base + off); off += align256((size_t)N * 128 * 2);
    float* agg = (float*)(base + off); off += align256((size_t)N * 128 * 4);
    _Float16* t2 = (_Float16*)(base + off); off += align256((size_t)N * 64 * 2);
    float* rs_out = deg;
    float* rs_in  = deg + N;

    const int TB = 256;
    zero_f32_kernel<<<1024, TB, 0, stream>>>(deg, (long long)2 * N);
    zero_f32_kernel<<<4096, TB, 0, stream>>>(agg, (long long)N * 128);
    degree_kernel<<<(E + TB - 1) / TB, TB, 0, stream>>>(src, dst, rs_out, rs_in, E);
    rsqrt_kernel<<<(2 * N + TB - 1) / TB, TB, 0, stream>>>(deg, 2 * N);
    // layer 1
    gcn_gemm_kernel<128, false><<<(N + 63) / 64, 256, 0, stream>>>(x, rs_out, W1, t1, N);
    {
        long long total = (long long)E * 32;
        agg_kernel<128><<<(int)((total + TB - 1) / TB), TB, 0, stream>>>(t1, src, dst, agg, E);
    }
    post1_kernel<<<(int)(((long long)N * 128 + TB - 1) / TB), TB, 0, stream>>>(
        agg, rs_out, rs_in, b1, t1, N);
    // layer 2
    zero_f32_kernel<<<4096, TB, 0, stream>>>(agg, (long long)N * 64);
    gcn_gemm_kernel<64, true><<<(N + 63) / 64, 256, 0, stream>>>(t1, nullptr, W2, t2, N);
    {
        long long total = (long long)E * 16;
        agg_kernel<64><<<(int)((total + TB - 1) / TB), TB, 0, stream>>>(t2, src, dst, agg, E);
    }
    post2_kernel<<<(int)(((long long)N * 64 + TB - 1) / TB), TB, 0, stream>>>(agg, rs_in, b2, N);
    // edge MLP + sigmoid
    edge_mlp_kernel<<<(E + 127) / 128, 256, 0, stream>>>(
        agg, src, dst, Wp1, bp1, Wp2, bp2, Wp3, bp3, out, E);
}
